// HEGNEncoder_51118700757530
// MI455X (gfx1250) — matrix operand order repaired and checked by reading, not compiled, not run on hardware
//
#include <hip/hip_runtime.h>
#include <hip/hip_bf16.h>
#include <math.h>

typedef _Float16 hlf;
typedef __attribute__((ext_vector_type(16))) _Float16 v16h;
typedef __attribute__((ext_vector_type(8)))  float    v8f;
typedef __attribute__((ext_vector_type(2)))  _Float16 h2;

#define NN 20000      // nodes
#define NE 320000     // edges
#define DD 128        // feature dim
#define EC 40000      // edge chunk
#define NL 3          // EGNN layers

#ifndef __has_builtin
#define __has_builtin(x) 0
#endif
#if __has_builtin(__builtin_amdgcn_tensor_load_to_lds)
#define GEMM_HAVE_TDM 1
#endif

// ---------------------------------------------------------------------------
// float <-> ordered-uint key for atomic float max
// ---------------------------------------------------------------------------
__device__ __forceinline__ unsigned fkey(float f) {
  unsigned u = __float_as_uint(f);
  return (u & 0x80000000u) ? ~u : (u | 0x80000000u);
}
__device__ __forceinline__ float fkey_inv(unsigned k) {
  unsigned u = (k & 0x80000000u) ? (k & 0x7fffffffu) : ~k;
  return __uint_as_float(u);
}

// ---------------------------------------------------------------------------
// TDM: DMA the packed weight panel (K/2 rows x 128 cols of 4B h2) into LDS.
// Descriptor layout per CDNA5 ISA ch.8 (group0 128b, group1 256b; 2D tile).
// ---------------------------------------------------------------------------
#ifdef GEMM_HAVE_TDM
typedef unsigned tdm_u4 __attribute__((ext_vector_type(4)));
typedef int      tdm_i8 __attribute__((ext_vector_type(8)));
typedef int      tdm_i4 __attribute__((ext_vector_type(4)));

__device__ __forceinline__ void tdm_load_panel(const h2* Bp, unsigned lds_off, int K) {
  unsigned long long ga = (unsigned long long)Bp;
  unsigned rows = (unsigned)(K >> 1);
  tdm_u4 g0;
  g0[0] = 1u;                                   // count=1, user mode, no gather
  g0[1] = lds_off;                              // lds_addr (bytes)
  g0[2] = (unsigned)ga;                         // global_addr[31:0]
  g0[3] = (unsigned)(ga >> 32) | (2u << 30);    // global_addr[56:32] | type=2
  tdm_i8 g1;
  g1[0] = (int)(2u << 16);                      // workgroup_mask=0, data_size=4B
  g1[1] = (int)(128u << 16);                    // tensor_dim0=128 (low16 @ bit48)
  g1[2] = (int)((rows & 0xFFFFu) << 16);        // td0 hi=0 | tensor_dim1 low16
  g1[3] = (int)((rows >> 16) | (128u << 16));   // td1 hi | tile_dim0=128
  g1[4] = (int)rows;                            // tile_dim1=rows, tile_dim2=0
  g1[5] = 128;                                  // tensor_dim0_stride low32
  g1[6] = 0;                                    // stride hi | dim1_stride low
  g1[7] = 0;
  tdm_i4 z4; z4[0] = z4[1] = z4[2] = z4[3] = 0;
#if __clang_major__ >= 23
  tdm_i8 z8; z8[0]=z8[1]=z8[2]=z8[3]=z8[4]=z8[5]=z8[6]=z8[7]=0;
  __builtin_amdgcn_tensor_load_to_lds(g0, g1, z4, z4, z8, 0);
#else
  __builtin_amdgcn_tensor_load_to_lds(g0, g1, z4, z4, 0);
#endif
}
__device__ __forceinline__ void tdm_wait() {
#if __has_builtin(__builtin_amdgcn_s_wait_tensorcnt)
  __builtin_amdgcn_s_wait_tensorcnt(0);
#else
  asm volatile("s_wait_tensorcnt 0" ::: "memory");
#endif
}
#endif

// ---------------------------------------------------------------------------
// WMMA GEMM: C[M,128] = act(A[M,K](f16) * Bpack[K,128](f16) + bias) (+resid)
// Weight panel staged in LDS once per workgroup (TDM if available).
// One wave -> 16 rows x full 128 cols.  K multiple of 32, M multiple of 16.
// Bpack layout: h2 Bpack[(k>>1)*128 + n] = { W[n][k], W[n][k+1] }  (B = W^T)
// ---------------------------------------------------------------------------
union AFrag { v16h v; h2 p[8]; uint4 q[2]; };
union CFrag { v8f v; float f[8]; };

template<int ACT> // 0 none, 1 silu, 2 relu
__global__ __launch_bounds__(128) void gemm_wmma_n128(
    const hlf* __restrict__ A, int lda, int K,
    const h2*  __restrict__ Bp,
    const float* __restrict__ bias,
    const float* __restrict__ resid,
    float* __restrict__ Cf, hlf* __restrict__ Ch, int M)
{
  extern __shared__ h2 Bs[];              // [K/2][128] h2  (K*256 bytes)
  const int wave  = threadIdx.x >> 5;
  const int lane  = threadIdx.x & 31;
  const int tileM = blockIdx.x * 4 + wave;

  // ---- stage weight panel into LDS (whole workgroup shares it) ----
#ifdef GEMM_HAVE_TDM
  if (wave == 0) {
    tdm_load_panel(Bp, (unsigned)(size_t)(&Bs[0]), K);
    tdm_wait();
  }
#else
  {
    const uint4* src = (const uint4*)Bp;
    uint4* dst = (uint4*)&Bs[0];
    int tot = K * 16;                     // K*256 bytes / 16
    for (int i = threadIdx.x; i < tot; i += 128) dst[i] = src[i];
  }
#endif
  __syncthreads();
  if (tileM * 16 >= M) return;

  const int hh = lane >> 4;     // lane half
  const int r  = lane & 15;     // A row within tile / B,C column
  const hlf* Arow = A + (size_t)(tileM * 16 + r) * lda;

  CFrag acc[8];
  #pragma unroll
  for (int nt = 0; nt < 8; ++nt) {
    float bv = bias ? bias[nt * 16 + r] : 0.0f;
    #pragma unroll
    for (int v = 0; v < 8; ++v) acc[nt].f[v] = bv;
  }

  for (int k0 = 0; k0 < K; k0 += 32) {
    AFrag a;   // two contiguous 16B runs per lane (k = 16*hh + s, same order as B)
    a.q[0] = *reinterpret_cast<const uint4*>(Arow + k0 + hh * 16);
    a.q[1] = *reinterpret_cast<const uint4*>(Arow + k0 + hh * 16 + 8);
    if (k0 + 64 < K)
      __builtin_prefetch(Arow + k0 + 64, 0, 3);   // global_prefetch_b8
    const int bbase = ((k0 >> 1) + hh * 8) * 128 + r;
    #pragma unroll
    for (int nt = 0; nt < 8; ++nt) {
      AFrag b;
      #pragma unroll
      for (int v = 0; v < 8; ++v) b.p[v] = Bs[bbase + v * 128 + nt * 16];
      acc[nt].v = __builtin_amdgcn_wmma_f32_16x16x32_f16(
          false, a.v, false, b.v, (short)0, acc[nt].v, false, false);
    }
  }

  #pragma unroll
  for (int nt = 0; nt < 8; ++nt) {
    #pragma unroll
    for (int v = 0; v < 8; ++v) {
      int m = hh * 8 + v;
      size_t idx = (size_t)(tileM * 16 + m) * 128 + nt * 16 + r;
      float x = acc[nt].f[v];
      if (ACT == 1) x = x / (1.0f + __expf(-x));
      if (ACT == 2) x = fmaxf(x, 0.0f);
      if (resid) x += resid[idx];
      if (Cf) Cf[idx] = x;
      if (Ch) Ch[idx] = (hlf)x;
    }
  }
}

// ---------------------------------------------------------------------------
// weight pack: W[128,K] f32 -> h2 Bpack[Kpad/2][128], zero-padded K..Kpad
// ---------------------------------------------------------------------------
union H2U { h2 v; hlf e[2]; };
__global__ void k_pack_weight(const float* __restrict__ W, h2* __restrict__ dst,
                              int K, int Kpad)
{
  int idx = blockIdx.x * blockDim.x + threadIdx.x;
  int total = 128 * (Kpad >> 1);
  if (idx >= total) return;
  int kp = idx / 128, n = idx - kp * 128;
  int k = kp * 2;
  float a = (k     < K) ? W[(size_t)n * K + k]     : 0.0f;
  float b = (k + 1 < K) ? W[(size_t)n * K + k + 1] : 0.0f;
  H2U u; u.e[0] = (hlf)a; u.e[1] = (hlf)b;
  dst[(size_t)kp * 128 + n] = u.v;
}

__global__ void k_f32_to_f16(hlf* dst, const float* src, int n) {
  int i = blockIdx.x * blockDim.x + threadIdx.x;
  if (i < n) dst[i] = (hlf)src[i];
}

// ---------------------------------------------------------------------------
// edge input build: Ae[e,0:288] = [h16[row], h16[col], radial, edge_attr, pad]
// ---------------------------------------------------------------------------
__global__ __launch_bounds__(128) void k_build_edge_input(
    hlf* __restrict__ Ae, const hlf* __restrict__ h16, const float* __restrict__ x,
    const float* __restrict__ ea, const int* __restrict__ row,
    const int* __restrict__ col, int e0)
{
  int el = blockIdx.x, t = threadIdx.x;
  int e = e0 + el;
  int rr = row[e], cc = col[e];
  size_t base = (size_t)el * 288;
  Ae[base + t]       = h16[(size_t)rr * 128 + t];
  Ae[base + 128 + t] = h16[(size_t)cc * 128 + t];
  if (t == 0) {
    float dx = x[rr*3+0]-x[cc*3+0], dy = x[rr*3+1]-x[cc*3+1], dz = x[rr*3+2]-x[cc*3+2];
    Ae[base + 256] = (hlf)(dx*dx + dy*dy + dz*dz);
  }
  if (t >= 1 && t < 17)  Ae[base + 256 + t] = (hlf)ea[(size_t)e * 16 + (t - 1)];
  if (t >= 17 && t < 32) Ae[base + 256 + t] = (hlf)0.0f;
}

// attn edge build: Ae2[e,0:256] = [hi_tan16[row], hi_tan16[col]]
__global__ __launch_bounds__(256) void k_build_attn_edge(
    hlf* __restrict__ Ae, const hlf* __restrict__ hi16,
    const int* __restrict__ row, const int* __restrict__ col, int e0)
{
  int el = blockIdx.x, t = threadIdx.x;
  int e = e0 + el;
  size_t base = (size_t)el * 256;
  if (t < 128) Ae[base + t] = hi16[(size_t)row[e] * 128 + t];
  else         Ae[base + t] = hi16[(size_t)col[e] * 128 + (t - 128)];
}

// node input build: An[n,0:256] = [h16, agg]
__global__ __launch_bounds__(256) void k_build_node_input(
    hlf* __restrict__ An, const hlf* __restrict__ h16, const float* __restrict__ agg)
{
  int n = blockIdx.x, t = threadIdx.x;
  size_t b = (size_t)n * 256;
  if (t < 128) An[b + t] = h16[(size_t)n * 128 + t];
  else         An[b + t] = (hlf)agg[(size_t)n * 128 + (t - 128)];
}

// ---------------------------------------------------------------------------
// per-edge dot products (wave per edge, 128 elems)
// ---------------------------------------------------------------------------
__global__ __launch_bounds__(128) void k_phi_coord(
    const hlf* __restrict__ t16, const float* __restrict__ cW2,
    const float* __restrict__ x, const int* __restrict__ row,
    const int* __restrict__ col, float* __restrict__ xacc, int e0, int ec)
{
  int wave = threadIdx.x >> 5, lane = threadIdx.x & 31;
  int el = blockIdx.x * 4 + wave;
  if (el >= ec) return;
  const hlf* t = t16 + (size_t)el * 128;
  float s = 0.0f;
  #pragma unroll
  for (int i = 0; i < 4; ++i) { int d = lane + 32*i; s += (float)t[d] * cW2[d]; }
  for (int off = 16; off; off >>= 1) s += __shfl_down(s, off);
  if (lane == 0) {
    int e = e0 + el, rr = row[e], cc = col[e];
    #pragma unroll
    for (int c = 0; c < 3; ++c) {
      float diff = x[rr*3+c] - x[cc*3+c];
      atomicAdd(&xacc[rr*3+c], diff * s);
    }
  }
}

__global__ __launch_bounds__(128) void k_edge_wscal(
    const hlf* __restrict__ w16, const float* __restrict__ aW2,
    const float* __restrict__ ab2, float* __restrict__ wscal, int e0, int ec)
{
  int wave = threadIdx.x >> 5, lane = threadIdx.x & 31;
  int el = blockIdx.x * 4 + wave;
  if (el >= ec) return;
  const hlf* t = w16 + (size_t)el * 128;
  float s = 0.0f;
  #pragma unroll
  for (int i = 0; i < 4; ++i) { int d = lane + 32*i; s += (float)t[d] * aW2[d]; }
  for (int off = 16; off; off >>= 1) s += __shfl_down(s, off);
  if (lane == 0) wscal[e0 + el] = s + ab2[0];
}

// ---------------------------------------------------------------------------
// scatters
// ---------------------------------------------------------------------------
__global__ void k_deg(const int* __restrict__ row, int* __restrict__ deg) {
  int e = blockIdx.x * blockDim.x + threadIdx.x;
  if (e < NE) atomicAdd(&deg[row[e]], 1);
}
__global__ void k_scatter_add_m(const hlf* __restrict__ m16, const int* __restrict__ row,
                                float* __restrict__ agg, int e0) {
  int idx = blockIdx.x * blockDim.x + threadIdx.x;
  if (idx >= EC * 128) return;
  int el = idx >> 7, d = idx & 127;
  atomicAdd(&agg[(size_t)row[e0 + el] * 128 + d], (float)m16[(size_t)el * 128 + d]);
}
__global__ void k_coord_update(float* __restrict__ x, const float* __restrict__ xacc,
                               const int* __restrict__ deg) {
  int i = blockIdx.x * blockDim.x + threadIdx.x;
  if (i < NN * 3) x[i] += xacc[i] / fmaxf((float)deg[i / 3], 1.0f);
}
__global__ void k_segmax(const float* __restrict__ w, const int* __restrict__ row,
                         unsigned* __restrict__ mx) {
  int e = blockIdx.x * blockDim.x + threadIdx.x;
  if (e < NE) atomicMax(&mx[row[e]], fkey(w[e]));
}
__global__ void k_segexp(const float* __restrict__ w, const int* __restrict__ row,
                         const unsigned* __restrict__ mx, float* __restrict__ aexp,
                         float* __restrict__ ssum) {
  int e = blockIdx.x * blockDim.x + threadIdx.x;
  if (e >= NE) return;
  int r = row[e];
  float a = __expf(w[e] - fkey_inv(mx[r]));
  aexp[e] = a;
  atomicAdd(&ssum[r], a);
}
__global__ void k_attn_scatter(const float* __restrict__ ht, const float* __restrict__ aexp,
                               const float* __restrict__ ssum, const int* __restrict__ row,
                               const int* __restrict__ col, float* __restrict__ agg) {
  long long idx = (long long)blockIdx.x * blockDim.x + threadIdx.x;
  if (idx >= (long long)NE * 128) return;
  int e = (int)(idx >> 7), d = (int)(idx & 127);
  int r = row[e];
  float coef = aexp[e] / ssum[r];
  atomicAdd(&agg[(size_t)r * 128 + d], ht[(size_t)col[e] * 128 + d] * coef);
}
__global__ void k_divide_deg(float* __restrict__ agg, const int* __restrict__ deg) {
  int idx = blockIdx.x * blockDim.x + threadIdx.x;
  if (idx < NN * 128) agg[idx] /= fmaxf((float)deg[idx >> 7], 1.0f);
}

// ---------------------------------------------------------------------------
// rowwise ops (wave per row of 128)
// ---------------------------------------------------------------------------
__global__ __launch_bounds__(128) void k_logmap(
    const float* __restrict__ src, const float* __restrict__ kap,
    float* __restrict__ dstf, hlf* __restrict__ dsth, int n)
{
  int wave = threadIdx.x >> 5, lane = threadIdx.x & 31;
  int rowi = blockIdx.x * 4 + wave;
  if (rowi >= n) return;
  size_t base = (size_t)rowi * 128;
  float v[4], s = 0.0f;
  #pragma unroll
  for (int i = 0; i < 4; ++i) { v[i] = src[base + lane + 32*i]; s += v[i]*v[i]; }
  for (int off = 16; off; off >>= 1) s += __shfl_down(s, off);
  s = __shfl(s, 0);
  float nn = fmaxf(sqrtf(s), 1e-5f);
  float kk = fminf(kap[0], -1e-5f), sk = sqrtf(-kk);
  float sc = atanhf(fminf(nn, 1.0f - 1e-5f)) / (sk * nn);
  #pragma unroll
  for (int i = 0; i < 4; ++i) {
    float o = v[i] * sc;
    int d = lane + 32*i;
    if (dstf) dstf[base + d] = o;
    if (dsth) dsth[base + d] = (hlf)o;
  }
}
__global__ __launch_bounds__(128) void k_expmap(
    const float* __restrict__ src, const float* __restrict__ kap,
    float* __restrict__ dstf, int n)
{
  int wave = threadIdx.x >> 5, lane = threadIdx.x & 31;
  int rowi = blockIdx.x * 4 + wave;
  if (rowi >= n) return;
  size_t base = (size_t)rowi * 128;
  float v[4], s = 0.0f;
  #pragma unroll
  for (int i = 0; i < 4; ++i) { v[i] = src[base + lane + 32*i]; s += v[i]*v[i]; }
  for (int off = 16; off; off >>= 1) s += __shfl_down(s, off);
  s = __shfl(s, 0);
  float nn = fmaxf(sqrtf(s), 1e-5f);
  float kk = fminf(kap[0], -1e-5f), sk = sqrtf(-kk);
  float sc = tanhf(sk * nn) / (sk * nn);
  #pragma unroll
  for (int i = 0; i < 4; ++i) dstf[base + lane + 32*i] = v[i] * sc;
}
__global__ __launch_bounds__(128) void k_ln_add(
    const float* __restrict__ a, const float* __restrict__ b,
    const float* __restrict__ g, const float* __restrict__ be,
    float* __restrict__ out, int n)
{
  int wave = threadIdx.x >> 5, lane = threadIdx.x & 31;
  int rowi = blockIdx.x * 4 + wave;
  if (rowi >= n) return;
  size_t base = (size_t)rowi * 128;
  float v[4], s = 0.0f;
  #pragma unroll
  for (int i = 0; i < 4; ++i) { v[i] = a[base+lane+32*i] + b[base+lane+32*i]; s += v[i]; }
  for (int off = 16; off; off >>= 1) s += __shfl_down(s, off);
  float mu = __shfl(s, 0) * (1.0f/128.0f);
  float q = 0.0f;
  #pragma unroll
  for (int i = 0; i < 4; ++i) { float d = v[i]-mu; q += d*d; }
  for (int off = 16; off; off >>= 1) q += __shfl_down(q, off);
  float inv = rsqrtf(__shfl(q, 0) * (1.0f/128.0f) + 1e-5f);
  #pragma unroll
  for (int i = 0; i < 4; ++i) {
    int d = lane + 32*i;
    out[base + d] = (v[i]-mu)*inv*g[d] + be[d];
  }
}
__global__ __launch_bounds__(128) void k_rownorm(
    float* __restrict__ xf, hlf* __restrict__ xh, int n)
{
  int wave = threadIdx.x >> 5, lane = threadIdx.x & 31;
  int rowi = blockIdx.x * 4 + wave;
  if (rowi >= n) return;
  size_t base = (size_t)rowi * 128;
  float v[4], s = 0.0f;
  #pragma unroll
  for (int i = 0; i < 4; ++i) { v[i] = xf[base + lane + 32*i]; s += v[i]*v[i]; }
  for (int off = 16; off; off >>= 1) s += __shfl_down(s, off);
  float nrm = fmaxf(sqrtf(__shfl(s, 0)), 1e-12f);
  #pragma unroll
  for (int i = 0; i < 4; ++i) {
    int d = lane + 32*i;
    float o = v[i] / nrm;
    xf[base + d] = o;
    xh[base + d] = (hlf)o;
  }
}

// ---------------------------------------------------------------------------
// global attention pooling
// ---------------------------------------------------------------------------
__global__ void k_qlin(const float* tok, const float* Wq, const float* bq, float* q) {
  int j = threadIdx.x;
  float s = bq[j];
  for (int k = 0; k < 128; ++k) s += tok[k] * Wq[j * 128 + k];
  q[j] = s;
}
__global__ void k_scores(const float* __restrict__ q, const float* __restrict__ kv,
                         float* __restrict__ sc) {
  int idx = blockIdx.x * blockDim.x + threadIdx.x;
  if (idx >= NN * 4) return;
  int nd = idx >> 2, h = idx & 3;
  float s = 0.0f;
  for (int d = 0; d < 32; ++d) s += q[h*32+d] * kv[(size_t)nd*128 + h*32 + d];
  sc[idx] = s * 0.17677669529663687f; // 1/sqrt(32)
}
__global__ void k_headmax(const float* sc, unsigned* smax) {
  int idx = blockIdx.x * blockDim.x + threadIdx.x;
  if (idx < NN * 4) atomicMax(&smax[idx & 3], fkey(sc[idx]));
}
__global__ void k_headexp(const float* sc, const unsigned* smax, float* esc, float* ssum) {
  int idx = blockIdx.x * blockDim.x + threadIdx.x;
  if (idx >= NN * 4) return;
  float e = __expf(sc[idx] - fkey_inv(smax[idx & 3]));
  esc[idx] = e;
  atomicAdd(&ssum[idx & 3], e);
}
__global__ __launch_bounds__(128) void k_accum_o(const float* __restrict__ esc,
                                                 const float* __restrict__ v,
                                                 float* __restrict__ o) {
  int d = threadIdx.x, h = d >> 5;
  float s = 0.0f;
  int base = blockIdx.x * 64;
  for (int i = 0; i < 64; ++i) {
    int nd = base + i;
    if (nd < NN) s += esc[nd * 4 + h] * v[(size_t)nd * 128 + d];
  }
  atomicAdd(&o[d], s);
}
__global__ void k_gr(const float* o, const float* ssum, const float* Wo,
                     const float* bo, float* out) {
  __shared__ float gr[128];
  __shared__ float red[128];
  int j = threadIdx.x;
  float s = bo[j];
  for (int k = 0; k < 128; ++k) s += (o[k] / ssum[k >> 5]) * Wo[j * 128 + k];
  gr[j] = s; red[j] = s * s;
  __syncthreads();
  for (int off = 64; off; off >>= 1) { if (j < off) red[j] += red[j + off]; __syncthreads(); }
  float nrm = fmaxf(sqrtf(red[0]), 1e-12f);
  out[j] = gr[j] / nrm;
}

// ---------------------------------------------------------------------------
// host orchestration
// ---------------------------------------------------------------------------
static void launch_gemm(int act, const hlf* A, int lda, int K, const h2* B,
                        const float* bias, const float* resid, float* Cf, hlf* Ch,
                        int M, hipStream_t s) {
  int blocks = (M / 16 + 3) >> 2;
  size_t shmem = (size_t)K * 256;   // weight panel bytes in LDS
  if (act == 1)      gemm_wmma_n128<1><<<blocks, 128, shmem, s>>>(A, lda, K, B, bias, resid, Cf, Ch, M);
  else if (act == 2) gemm_wmma_n128<2><<<blocks, 128, shmem, s>>>(A, lda, K, B, bias, resid, Cf, Ch, M);
  else               gemm_wmma_n128<0><<<blocks, 128, shmem, s>>>(A, lda, K, B, bias, resid, Cf, Ch, M);
}
static void launch_pack(const float* W, h2* dst, int K, int Kpad, hipStream_t s) {
  int total = 128 * (Kpad >> 1);
  k_pack_weight<<<(total + 255) / 256, 256, 0, s>>>(W, dst, K, Kpad);
}

extern "C" void kernel_launch(void* const* d_in, const int* in_sizes, int n_in,
                              void* d_out, int out_size, void* d_ws, size_t ws_size,
                              hipStream_t stream) {
  (void)in_sizes; (void)n_in; (void)out_size; (void)ws_size;
  const float* H        = (const float*)d_in[0];
  const float* Z        = (const float*)d_in[1];
  const int*   edges    = (const int*)d_in[2];
  const float* ea       = (const float*)d_in[3];
  const float* emb_in_W = (const float*)d_in[4];  const float* emb_in_b = (const float*)d_in[5];
  const float* emb_out_W= (const float*)d_in[6];  const float* emb_out_b= (const float*)d_in[7];
  const float* eW1 = (const float*)d_in[8];   const float* eb1 = (const float*)d_in[9];
  const float* eW2 = (const float*)d_in[10];  const float* eb2 = (const float*)d_in[11];
  const float* nW1 = (const float*)d_in[12];  const float* nb1 = (const float*)d_in[13];
  const float* nW2 = (const float*)d_in[14];  const float* nb2 = (const float*)d_in[15];
  const float* cW1 = (const float*)d_in[16];  const float* cb1 = (const float*)d_in[17];
  const float* cW2 = (const float*)d_in[18];
  const float* e2h_W = (const float*)d_in[19]; const float* e2h_b = (const float*)d_in[20];
  const float* e2h_k = (const float*)d_in[21];
  const float* hgc_k = (const float*)d_in[22];
  const float* hl_W  = (const float*)d_in[23]; const float* hl_b = (const float*)d_in[24];
  const float* hl_k  = (const float*)d_in[25];
  const float* attn_k= (const float*)d_in[26];
  const float* aW1   = (const float*)d_in[27]; const float* ab1 = (const float*)d_in[28];
  const float* aW2   = (const float*)d_in[29]; const float* ab2 = (const float*)d_in[30];
  const float* ln_g  = (const float*)d_in[31]; const float* ln_b = (const float*)d_in[32];
  const float* rp_W  = (const float*)d_in[33]; const float* rp_b = (const float*)d_in[34];
  const float* fp_W  = (const float*)d_in[35]; const float* fp_b = (const float*)d_in[36];
  const float* Wq = (const float*)d_in[37];  const float* bq = (const float*)d_in[38];
  const float* Wk = (const float*)d_in[39];  const float* bk = (const float*)d_in[40];
  const float* Wv = (const float*)d_in[41];  const float* bv = (const float*)d_in[42];
  const float* Wo = (const float*)d_in[43];  const float* bo = (const float*)d_in[44];
  const float* gtok = (const float*)d_in[45];

  const int* row = edges;
  const int* col = edges + NE;

  // ---- workspace carve ----
  char* p = (char*)d_ws;
  auto alloc = [&](size_t bytes) -> void* {
    void* r = (void*)p; p += (bytes + 255) & ~(size_t)255; return r;
  };
  float* h     = (float*)alloc((size_t)NN * 128 * 4);
  hlf*   h16   = (hlf*)  alloc((size_t)NN * 128 * 2);
  float* agg   = (float*)alloc((size_t)NN * 128 * 4);
  float* x     = (float*)alloc((size_t)NN * 3 * 4);
  float* xacc  = (float*)alloc((size_t)NN * 3 * 4);
  int*   deg   = (int*)  alloc((size_t)NN * 4);
  hlf*   Ae    = (hlf*)  alloc((size_t)EC * 288 * 2);
  hlf*   bufB  = (hlf*)  alloc((size_t)EC * 128 * 2);
  hlf*   bufC  = (hlf*)  alloc((size_t)EC * 128 * 2);
  hlf*   An16  = (hlf*)  alloc((size_t)NN * 256 * 2);
  hlf*   t16   = (hlf*)  alloc((size_t)NN * 128 * 2);
  float* tmpA  = (float*)alloc((size_t)NN * 128 * 4);
  float* tmpB  = (float*)alloc((size_t)NN * 128 * 4);
  float* tmpC  = (float*)alloc((size_t)NN * 128 * 4);
  float* wscal = (float*)alloc((size_t)NE * 4);
  float* aexp  = (float*)alloc((size_t)NE * 4);
  unsigned* segmax = (unsigned*)alloc((size_t)NN * 4);
  float* segsum = (float*)alloc((size_t)NN * 4);
  float* kv    = (float*)alloc((size_t)NN * 128 * 4);
  float* vbuf  = (float*)alloc((size_t)NN * 128 * 4);
  float* scores= (float*)alloc((size_t)NN * 4 * 4);
  float* esc   = (float*)alloc((size_t)NN * 4 * 4);
  float* qv    = (float*)alloc(128 * 4);
  float* ovec  = (float*)alloc(128 * 4);
  unsigned* smax = (unsigned*)alloc(4 * 4);
  float* ssum  = (float*)alloc(4 * 4);
  // weight packs
  h2* p_emb_in  = (h2*)alloc((size_t)64  * 128 * 4);
  h2* p_emb_out = (h2*)alloc((size_t)64  * 128 * 4);
  h2* p_eW1     = (h2*)alloc((size_t)3 * 144 * 128 * 4);
  h2* p_eW2     = (h2*)alloc((size_t)3 * 64  * 128 * 4);
  h2* p_nW1     = (h2*)alloc((size_t)3 * 128 * 128 * 4);
  h2* p_nW2     = (h2*)alloc((size_t)3 * 64  * 128 * 4);
  h2* p_cW1     = (h2*)alloc((size_t)3 * 64  * 128 * 4);
  h2* p_e2h     = (h2*)alloc((size_t)64  * 128 * 4);
  h2* p_hl      = (h2*)alloc((size_t)64  * 128 * 4);
  h2* p_aW1     = (h2*)alloc((size_t)128 * 128 * 4);
  h2* p_rp      = (h2*)alloc((size_t)64  * 128 * 4);
  h2* p_fp      = (h2*)alloc((size_t)64  * 128 * 4);
  h2* p_Wk      = (h2*)alloc((size_t)64  * 128 * 4);
  h2* p_Wv      = (h2*)alloc((size_t)64  * 128 * 4);

  // ---- weight conversion ----
  launch_pack(emb_in_W,  p_emb_in,  128, 128, stream);
  launch_pack(emb_out_W, p_emb_out, 128, 128, stream);
  for (int l = 0; l < NL; ++l) {
    launch_pack(eW1 + (size_t)l*128*273, p_eW1 + (size_t)l*144*128, 273, 288, stream);
    launch_pack(eW2 + (size_t)l*128*128, p_eW2 + (size_t)l*64*128,  128, 128, stream);
    launch_pack(nW1 + (size_t)l*128*256, p_nW1 + (size_t)l*128*128, 256, 256, stream);
    launch_pack(nW2 + (size_t)l*128*128, p_nW2 + (size_t)l*64*128,  128, 128, stream);
    launch_pack(cW1 + (size_t)l*128*128, p_cW1 + (size_t)l*64*128,  128, 128, stream);
  }
  launch_pack(e2h_W, p_e2h, 128, 128, stream);
  launch_pack(hl_W,  p_hl,  128, 128, stream);
  launch_pack(aW1,   p_aW1, 256, 256, stream);
  launch_pack(rp_W,  p_rp,  128, 128, stream);
  launch_pack(fp_W,  p_fp,  128, 128, stream);
  launch_pack(Wk,    p_Wk,  128, 128, stream);
  launch_pack(Wv,    p_Wv,  128, 128, stream);

  // ---- init ----
  hipMemcpyAsync(x, Z, (size_t)NN * 3 * 4, hipMemcpyDeviceToDevice, stream);
  hipMemsetAsync(deg, 0, (size_t)NN * 4, stream);
  k_deg<<<(NE + 255) / 256, 256, 0, stream>>>(row, deg);
  k_f32_to_f16<<<(NN * 128 + 255) / 256, 256, 0, stream>>>(t16, H, NN * 128);
  // h = lin(H, emb_in)
  launch_gemm(0, t16, 128, 128, p_emb_in, emb_in_b, nullptr, h, h16, NN, stream);

  // ---- EGNN layers ----
  for (int l = 0; l < NL; ++l) {
    hipMemsetAsync(xacc, 0, (size_t)NN * 3 * 4, stream);
    hipMemsetAsync(agg,  0, (size_t)NN * 128 * 4, stream);
    for (int c = 0; c < NE / EC; ++c) {
      int e0 = c * EC;
      k_build_edge_input<<<EC, 128, 0, stream>>>(Ae, h16, x, ea, row, col, e0);
      launch_gemm(1, Ae,   288, 288, p_eW1 + (size_t)l*144*128, eb1 + l*128, nullptr, nullptr, bufB, EC, stream);
      launch_gemm(1, bufB, 128, 128, p_eW2 + (size_t)l*64*128,  eb2 + l*128, nullptr, nullptr, bufC, EC, stream);
      launch_gemm(1, bufC, 128, 128, p_cW1 + (size_t)l*64*128,  cb1 + l*128, nullptr, nullptr, bufB, EC, stream);
      k_phi_coord<<<(EC + 3) / 4, 128, 0, stream>>>(bufB, cW2 + l*128, x, row, col, xacc, e0, EC);
      k_scatter_add_m<<<(EC * 128 + 255) / 256, 256, 0, stream>>>(bufC, row, agg, e0);
    }
    k_coord_update<<<(NN * 3 + 255) / 256, 256, 0, stream>>>(x, xacc, deg);
    k_build_node_input<<<NN, 256, 0, stream>>>(An16, h16, agg);
    launch_gemm(1, An16, 256, 256, p_nW1 + (size_t)l*128*128, nb1 + l*128, nullptr, nullptr, t16, NN, stream);
    launch_gemm(0, t16,  128, 128, p_nW2 + (size_t)l*64*128,  nb2 + l*128, h, h, h16, NN, stream);
  }
  // h = lin(h, emb_out)
  launch_gemm(0, h16, 128, 128, p_emb_out, emb_out_b, nullptr, h, nullptr, NN, stream);

  // ---- e2h: exp(lin(log(h))) ----
  int rb = (NN + 3) / 4;
  k_logmap<<<rb, 128, 0, stream>>>(h, e2h_k, nullptr, t16, NN);
  launch_gemm(0, t16, 128, 128, p_e2h, e2h_b, nullptr, tmpA, nullptr, NN, stream);
  k_expmap<<<rb, 128, 0, stream>>>(tmpA, e2h_k, h, NN);

  // ---- hyperbolic GC layer ----
  k_logmap<<<rb, 128, 0, stream>>>(h, hl_k, nullptr, t16, NN);
  launch_gemm(0, t16, 128, 128, p_hl, hl_b, nullptr, tmpA, nullptr, NN, stream);
  k_expmap<<<rb, 128, 0, stream>>>(tmpA, hl_k, tmpB, NN);             // ht = tmpB
  k_logmap<<<rb, 128, 0, stream>>>(tmpB, attn_k, nullptr, t16, NN);    // hi_tan16 = t16
  hipMemsetAsync(segmax, 0, (size_t)NN * 4, stream);
  hipMemsetAsync(segsum, 0, (size_t)NN * 4, stream);
  for (int c = 0; c < NE / EC; ++c) {
    int e0 = c * EC;
    k_build_attn_edge<<<EC, 256, 0, stream>>>(Ae, t16, row, col, e0);
    launch_gemm(2, Ae, 256, 256, p_aW1, ab1, nullptr, nullptr, bufB, EC, stream);
    k_edge_wscal<<<(EC + 3) / 4, 128, 0, stream>>>(bufB, aW2, ab2, wscal, e0, EC);
  }
  k_segmax<<<(NE + 255) / 256, 256, 0, stream>>>(wscal, row, segmax);
  k_segexp<<<(NE + 255) / 256, 256, 0, stream>>>(wscal, row, segmax, aexp, segsum);
  hipMemsetAsync(agg, 0, (size_t)NN * 128 * 4, stream);
  {
    long long tot = (long long)NE * 128;
    k_attn_scatter<<<(unsigned)((tot + 255) / 256), 256, 0, stream>>>(tmpB, aexp, segsum, row, col, agg);
  }
  k_divide_deg<<<(NN * 128 + 255) / 256, 256, 0, stream>>>(agg, deg);
  k_logmap<<<rb, 128, 0, stream>>>(agg, attn_k, tmpA, nullptr, NN);    // h_agg_tan
  k_logmap<<<rb, 128, 0, stream>>>(h, attn_k, nullptr, t16, NN);
  launch_gemm(0, t16, 128, 128, p_rp, rp_b, nullptr, tmpC, nullptr, NN, stream); // h_res_tan
  k_ln_add<<<rb, 128, 0, stream>>>(tmpA, tmpC, ln_g, ln_b, tmpB, NN);
  k_expmap<<<rb, 128, 0, stream>>>(tmpB, hgc_k, h, NN);

  // ---- feature projection + row normalize -> hf (first NN*128 of d_out) ----
  float* hf = (float*)d_out;
  k_logmap<<<rb, 128, 0, stream>>>(h, hgc_k, nullptr, t16, NN);
  launch_gemm(0, t16, 128, 128, p_fp, fp_b, nullptr, hf, nullptr, NN, stream);
  k_rownorm<<<rb, 128, 0, stream>>>(hf, t16, NN);                      // t16 = hf16

  // ---- multi-head attention pooling ----
  launch_gemm(0, t16, 128, 128, p_Wk, bk, nullptr, kv,   nullptr, NN, stream);
  launch_gemm(0, t16, 128, 128, p_Wv, bv, nullptr, vbuf, nullptr, NN, stream);
  k_qlin<<<1, 128, 0, stream>>>(gtok, Wq, bq, qv);
  k_scores<<<(NN * 4 + 255) / 256, 256, 0, stream>>>(qv, kv, scores);
  hipMemsetAsync(smax, 0, 4 * 4, stream);
  hipMemsetAsync(ssum, 0, 4 * 4, stream);
  hipMemsetAsync(ovec, 0, 128 * 4, stream);
  k_headmax<<<(NN * 4 + 255) / 256, 256, 0, stream>>>(scores, smax);
  k_headexp<<<(NN * 4 + 255) / 256, 256, 0, stream>>>(scores, smax, esc, ssum);
  k_accum_o<<<(NN + 63) / 64, 128, 0, stream>>>(esc, vbuf, ovec);
  k_gr<<<1, 128, 0, stream>>>(ovec, ssum, Wo, bo, hf + (size_t)NN * 128);
}
